// MultiHeadAttention_12128987644307
// MI455X (gfx1250) — matrix operand order repaired and checked by reading, not compiled, not run on hardware
//
#include <hip/hip_runtime.h>

// ---------------------------------------------------------------------------
// MultiHeadAttention for MI455X (gfx1250): bf16 WMMA GEMMs + flash attention.
// B=4, F=T=2048, D=1024, N=16 heads, H=64.
//
// Pipeline:
//   1) cvt_act_kernel:  from/to fp32 -> bf16 copies (one pass, bandwidth-bound)
//   2) cvt_wt_kernel:   wq/wk/wv fp32 [D,NH] -> bf16 TRANSPOSED [NH,D]
//   3) qkv_gemm_kernel: 32x64 tile/wave, 8 WMMAs + 12 b128 loads per K-step
//   4) flash_attn_kernel: streaming online softmax, T-step=64,
//        row-sums computed by WMMA against a constant ones matrix
//        (no shuffle-based sum reductions), probs never materialized.
// ---------------------------------------------------------------------------

#define B_     4
#define F_     2048
#define T_     2048
#define D_     1024
#define NHEAD  16
#define H_     64
#define NH_    1024                     // N*H
#define LOG2E  1.44269504088896340736f
#define QSCALE (0.125f * LOG2E)         // 1/sqrt(64) folded with log2(e)
#define MASKV  (-1.0e9f * LOG2E)

typedef __attribute__((ext_vector_type(16))) __bf16   v16bf;
typedef __attribute__((ext_vector_type(8)))  float    v8f;
typedef __attribute__((ext_vector_type(4)))  float    v4f;
typedef __attribute__((ext_vector_type(4)))  unsigned v4u;
typedef __attribute__((ext_vector_type(2)))  unsigned v2u;

union opbf { v4u q[2]; v16bf v; };      // one WMMA operand = 2 x b128 loads

// fp32 -> bf16 round-to-nearest-even (scalar; used only outside hot GEMM loops)
__device__ __forceinline__ __bf16 f2bf(float x) {
  union { float f; unsigned u; } a; a.f = x;
  unsigned r = a.u + 0x7FFFu + ((a.u >> 16) & 1u);
  union { unsigned short s; __bf16 b; } o;
  o.s = (unsigned short)(r >> 16);
  return o.b;
}
// pack two fp32 -> one dword of two bf16 (RNE)
__device__ __forceinline__ unsigned packbf2(float x, float y) {
  union { float f; unsigned u; } a, b; a.f = x; b.f = y;
  unsigned lo = (a.u + 0x7FFFu + ((a.u >> 16) & 1u)) >> 16;
  unsigned hi = (b.u + 0x7FFFu + ((b.u >> 16) & 1u)) & 0xFFFF0000u;
  return lo | hi;
}

// A operand (16x32 ISA layout): lane half hi holds K = {8hi..8hi+7, 16+8hi..16+8hi+7}
__device__ __forceinline__ v16bf ld_a(const __bf16* base, int hi) {
  opbf t;
  t.q[0] = *(const v4u*)(base + 8 * hi);
  t.q[1] = *(const v4u*)(base + 16 + 8 * hi);
  return t.v;
}
// B operand (32x16 ISA layout): lane half hi holds K = {8hi..8hi+7, 16+8hi..16+8hi+7}
__device__ __forceinline__ v16bf ld_b(const __bf16* base, int hi) {
  opbf t;
  t.q[0] = *(const v4u*)(base + 8 * hi);
  t.q[1] = *(const v4u*)(base + 16 + 8 * hi);
  return t.v;
}
// constant all-ones bf16 operand (for WMMA row-sum reductions)
__device__ __forceinline__ v16bf ones16() {
  opbf t;
#pragma unroll
  for (int i = 0; i < 4; ++i) { t.q[0][i] = 0x3F803F80u; t.q[1][i] = 0x3F803F80u; }
  return t.v;
}

__device__ __forceinline__ v8f wmma_bf16(v16bf a, v16bf b, v8f c) {
  return __builtin_amdgcn_wmma_f32_16x16x32_bf16(false, a, false, b,
                                                 (short)0, c, false, false);
}

// ---------------------------------------------------------------------------
// Kernel 1: activations fp32 -> bf16 (from_tensor and to_tensor), vectorized.
// ---------------------------------------------------------------------------
__global__ __launch_bounds__(256)
void cvt_act_kernel(const float* __restrict__ from_t, const float* __restrict__ to_t,
                    __bf16* __restrict__ fbf, __bf16* __restrict__ tbf) {
  const size_t n4 = (size_t)B_ * F_ * D_ / 4;   // float4 groups per tensor
  size_t t = (size_t)blockIdx.x * 256 + threadIdx.x;
  const float* src; __bf16* dst; size_t base;
  if (t < n4) { src = from_t; dst = fbf; base = t * 4; }
  else        { src = to_t;   dst = tbf; base = (t - n4) * 4; }
  v4f x = *(const v4f*)(src + base);
  v2u o; o[0] = packbf2(x[0], x[1]); o[1] = packbf2(x[2], x[3]);
  *(v2u*)(dst + base) = o;
}

// ---------------------------------------------------------------------------
// Kernel 2: weights fp32 [D,NH] -> bf16 transposed [NH,D] so the GEMM B-operand
// is lane-contiguous. One dword (2 bf16 along D) per thread.
// ---------------------------------------------------------------------------
__global__ __launch_bounds__(256)
void cvt_wt_kernel(const float* __restrict__ wq, const float* __restrict__ wk,
                   const float* __restrict__ wv,
                   __bf16* __restrict__ wtq, __bf16* __restrict__ wtk,
                   __bf16* __restrict__ wtv) {
  const size_t per = (size_t)NH_ * D_ / 2;      // dword outputs per weight
  size_t t = (size_t)blockIdx.x * 256 + threadIdx.x;
  int which = (int)(t / per);
  size_t rem = t - (size_t)which * per;
  int j  = (int)(rem / (D_ / 2));
  int dp = (int)(rem % (D_ / 2));
  const float* w = (which == 0) ? wq : (which == 1) ? wk : wv;
  __bf16*     wt = (which == 0) ? wtq : (which == 1) ? wtk : wtv;
  float x0 = w[(size_t)(2 * dp)     * NH_ + j];
  float x1 = w[(size_t)(2 * dp + 1) * NH_ + j];
  *(unsigned*)(wt + (size_t)j * D_ + 2 * dp) = packbf2(x0, x1);
}

// ---------------------------------------------------------------------------
// Kernel 3: QKV projections. One wave -> 32x64 output tile of X(2048xD)*W(DxNH).
// Per K-step(32): 2 A-operand loads + 4 B-operand loads -> 8 WMMAs.
// which=0: q -> qw [B,F,NH]; which=1: k -> kw [B,T,NH]; which=2: v -> vtw [B,NH,T]
// ---------------------------------------------------------------------------
__global__ __launch_bounds__(256)
void qkv_gemm_kernel(const __bf16* __restrict__ fbf, const __bf16* __restrict__ tbf,
                     const __bf16* __restrict__ wtq, const __bf16* __restrict__ wtk,
                     const __bf16* __restrict__ wtv,
                     const float* __restrict__ bq, const float* __restrict__ bk,
                     const float* __restrict__ bv,
                     __bf16* __restrict__ qw, __bf16* __restrict__ kw,
                     __bf16* __restrict__ vtw) {
  int g    = (blockIdx.x * blockDim.x + threadIdx.x) >> 5;
  int lane = threadIdx.x & 31;
  int nt = g & 15;            // 16 column tiles of 64
  int st = (g >> 4) & 63;     // 64 row tiles of 32
  int wb = g >> 10;           // 0..11 = (b, which)
  int which = wb % 3;
  int b     = wb / 3;

  const __bf16* X  = (which == 0) ? fbf : tbf;
  const __bf16* WT = (which == 0) ? wtq : (which == 1) ? wtk : wtv;
  const float* bias = (which == 0) ? bq : (which == 1) ? bk : bv;

  int m = lane & 15, hi = lane >> 4;
  int s0 = st * 32, j0 = nt * 64;

  const __bf16* xrow0 = X + ((size_t)b * F_ + s0 + m) * D_;
  const __bf16* xrow1 = xrow0 + (size_t)16 * D_;
  const __bf16* wrow0 = WT + (size_t)(j0 + m) * D_;

  v8f acc[2][4] = {};
  for (int d0 = 0; d0 < D_; d0 += 32) {
    v16bf a0 = ld_a(xrow0 + d0, hi);
    v16bf a1 = ld_a(xrow1 + d0, hi);
#pragma unroll
    for (int nb = 0; nb < 4; ++nb) {
      v16bf bm = ld_b(wrow0 + (size_t)(nb * 16) * D_ + d0, hi);
      acc[0][nb] = wmma_bf16(a0, bm, acc[0][nb]);
      acc[1][nb] = wmma_bf16(a1, bm, acc[1][nb]);
    }
  }

#pragma unroll
  for (int nb = 0; nb < 4; ++nb) {
    int jcol = j0 + nb * 16 + m;
    float bias_v = bias[jcol];
    if (which == 2) {
      __bf16* vp = vtw + ((size_t)b * NH_ + jcol) * T_;
#pragma unroll
      for (int ma = 0; ma < 2; ++ma)
#pragma unroll
        for (int r = 0; r < 8; ++r)
          vp[s0 + 16 * ma + r + 8 * hi] = f2bf(acc[ma][nb][r] + bias_v);
    } else {
      __bf16* outp = ((which == 0) ? qw : kw) + (size_t)b * F_ * NH_;
#pragma unroll
      for (int ma = 0; ma < 2; ++ma)
#pragma unroll
        for (int r = 0; r < 8; ++r)
          outp[(size_t)(s0 + 16 * ma + r + 8 * hi) * NH_ + jcol] =
              f2bf(acc[ma][nb][r] + bias_v);
    }
  }
}

// ---------------------------------------------------------------------------
// Kernel 4: flash attention. One wave owns (b, head, 16 F-rows); streams T in
// steps of 64. Base-2 online softmax; running denominator accumulated by
// WMMA against a constant ones matrix (acc_l) instead of lane shuffles.
// Per step: 8 QK^T WMMAs + 8 PV WMMAs + 2 sum WMMAs.
// ---------------------------------------------------------------------------
__global__ __launch_bounds__(256)
void flash_attn_kernel(const __bf16* __restrict__ qw, const __bf16* __restrict__ kw,
                       const __bf16* __restrict__ vtw, const int* __restrict__ amask,
                       float* __restrict__ out) {
  __shared__ __bf16 plds[8 * 16 * 64];   // per-wave 16x64 bf16 P staging (2KB)
  int g      = (blockIdx.x * blockDim.x + threadIdx.x) >> 5;
  int lane   = threadIdx.x & 31;
  int wlocal = threadIdx.x >> 5;
  __bf16* P  = plds + wlocal * (16 * 64);

  int ft = g & 127;           // 128 F tiles of 16 rows
  int n  = (g >> 7) & 15;     // head
  int b  = g >> 11;           // batch

  int m = lane & 15, hi = lane >> 4;
  int f0 = ft * 16;

  // Preload Q tile (16 x 64) in A layout: 2 chunks of K=32, 2 b128 loads each.
  const __bf16* qrow = qw + ((size_t)b * F_ + (f0 + m)) * NH_ + n * H_;
  v16bf aq0 = ld_a(qrow, hi);
  v16bf aq1 = ld_a(qrow + 32, hi);
  const v16bf vones = ones16();

  float mrun[8];
#pragma unroll
  for (int r = 0; r < 8; ++r) mrun[r] = -3.0e38f;
  v8f acc[4] = {};
  v8f acc_l = {};                        // running softmax denominator (all cols equal)

  const __bf16* kbase_p = kw + (size_t)b * T_ * NH_ + n * H_;
  const __bf16* vbase_p = vtw + ((size_t)b * NH_ + n * H_) * T_;
  const int* mrow = amask + b * T_;

  for (int t0 = 0; t0 < T_; t0 += 64) {
    // prefetch next iteration's K rows (lowers to global_prefetch_b8)
    if (t0 + 64 < T_)
      __builtin_prefetch(kbase_p + (size_t)(t0 + 64 + m) * NH_, 0, 1);

    // ---- scores: four 16x16 tiles over columns t0..t0+63 (K = H = 64) ----
    v8f s[4];
    float cmask[4];
#pragma unroll
    for (int cc = 0; cc < 4; ++cc) {
      int tcol = t0 + 16 * cc + m;
      cmask[cc] = (mrow[tcol] == 0) ? MASKV : 0.0f;
      const __bf16* krow = kbase_p + (size_t)tcol * NH_;
      v8f sc = {};
      sc = wmma_bf16(aq0, ld_b(krow, hi), sc);
      sc = wmma_bf16(aq1, ld_b(krow + 32, hi), sc);
      s[cc] = sc;
    }

    // ---- base-2 online softmax over the 64 new columns ----
    float p[4][8];
#pragma unroll
    for (int r = 0; r < 8; ++r) {
      float sv[4];
#pragma unroll
      for (int cc = 0; cc < 4; ++cc) sv[cc] = s[cc][r] * QSCALE + cmask[cc];
      float rmax = fmaxf(fmaxf(sv[0], sv[1]), fmaxf(sv[2], sv[3]));
#pragma unroll
      for (int off = 1; off < 16; off <<= 1)
        rmax = fmaxf(rmax, __shfl_xor(rmax, off, 32));   // within 16-lane half
      float mnew = fmaxf(mrun[r], rmax);
      float corr = exp2f(mrun[r] - mnew);
      mrun[r] = mnew;
#pragma unroll
      for (int cc = 0; cc < 4; ++cc) p[cc][r] = exp2f(sv[cc] - mnew);
      // rescale output accumulators AND the denominator accumulator
#pragma unroll
      for (int q4 = 0; q4 < 4; ++q4) acc[q4][r] *= corr;
      acc_l[r] *= corr;
    }

    // ---- re-layout P (C/D format -> A format) through per-wave LDS ----
#pragma unroll
    for (int cc = 0; cc < 4; ++cc)
#pragma unroll
      for (int r = 0; r < 8; ++r)
        P[(r + 8 * hi) * 64 + 16 * cc + m] = f2bf(p[cc][r]);
    asm volatile("s_wait_dscnt 0" ::: "memory");   // wave-local LDS RAW
    opbf t0p, t1p;                                 // A-format read: 4x ds_load_b128
    t0p.q[0] = *(const v4u*)(P + m * 64 + 8 * hi);
    t0p.q[1] = *(const v4u*)(P + m * 64 + 16 + 8 * hi);
    t1p.q[0] = *(const v4u*)(P + m * 64 + 32 + 8 * hi);
    t1p.q[1] = *(const v4u*)(P + m * 64 + 48 + 8 * hi);
    v16bf ap0 = t0p.v, ap1 = t1p.v;

    // ---- PV: out(16x64) += P(16x64) * V(64x64), 4 column tiles x K=64 ----
#pragma unroll
    for (int nhi = 0; nhi < 4; ++nhi) {
      const __bf16* vrow = vbase_p + (size_t)(nhi * 16 + m) * T_ + t0;
      acc[nhi] = wmma_bf16(ap0, ld_b(vrow, hi), acc[nhi]);
      acc[nhi] = wmma_bf16(ap1, ld_b(vrow + 32, hi), acc[nhi]);
    }
    // ---- denominator: acc_l += P * ones (row sums, replicated per column) ----
    acc_l = wmma_bf16(ap0, vones, acc_l);
    acc_l = wmma_bf16(ap1, vones, acc_l);
  }

  // ---- epilogue: normalize and store [B,F,N,H] fp32 ----
  float* obase = out + (((size_t)b * F_ + f0) * NHEAD + n) * H_;
#pragma unroll
  for (int r = 0; r < 8; ++r) {
    float inv = 1.0f / acc_l[r];
    int frow = r + 8 * hi;
#pragma unroll
    for (int nhi = 0; nhi < 4; ++nhi)
      obase[(size_t)frow * (NHEAD * H_) + nhi * 16 + m] = acc[nhi][r] * inv;
  }
}

// ---------------------------------------------------------------------------
extern "C" void kernel_launch(void* const* d_in, const int* in_sizes, int n_in,
                              void* d_out, int out_size, void* d_ws, size_t ws_size,
                              hipStream_t stream) {
  const float* from_t = (const float*)d_in[0];
  const float* to_t   = (const float*)d_in[1];
  const int*   amask  = (const int*)d_in[2];
  const float* wq = (const float*)d_in[3];
  const float* bq = (const float*)d_in[4];
  const float* wk = (const float*)d_in[5];
  const float* bk = (const float*)d_in[6];
  const float* wv = (const float*)d_in[7];
  const float* bv = (const float*)d_in[8];

  // workspace (bf16 elems): qkv (3x8M) + act copies (2x8M) + wT (3x1M) ~ 86MB
  const size_t e8m = (size_t)B_ * F_ * NH_;     // 8,388,608
  const size_t e1m = (size_t)NH_ * D_;          // 1,048,576
  __bf16* qw  = (__bf16*)d_ws;
  __bf16* kw  = qw + e8m;
  __bf16* vtw = kw + e8m;
  __bf16* fbf = vtw + e8m;
  __bf16* tbf = fbf + e8m;
  __bf16* wtq = tbf + e8m;
  __bf16* wtk = wtq + e1m;
  __bf16* wtv = wtk + e1m;

  // 1) activation conversion: 2 * 8M / 4 = 4M threads
  cvt_act_kernel<<<dim3(16384), dim3(256), 0, stream>>>(from_t, to_t, fbf, tbf);
  // 2) weight conversion+transpose: 3 * 1M / 2 = 1.5M threads
  cvt_wt_kernel<<<dim3(6144), dim3(256), 0, stream>>>(wq, wk, wv, wtq, wtk, wtv);
  // 3) QKV GEMMs: B*3*(2048/32)*(1024/64) = 12288 waves, 8 waves/block
  qkv_gemm_kernel<<<dim3(1536), dim3(256), 0, stream>>>(
      fbf, tbf, wtq, wtk, wtv, bq, bk, bv, qw, kw, vtw);
  // 4) flash attention: B*N*(F/16) = 8192 waves, 8 waves/block
  flash_attn_kernel<<<dim3(1024), dim3(256), 0, stream>>>(qw, kw, vtw, amask,
                                                          (float*)d_out);
}
